// AttentionLayer_83081847374008
// MI455X (gfx1250) — matrix operand order repaired and checked by reading, not compiled, not run on hardware
//
#include <hip/hip_runtime.h>
#include <hip/hip_bf16.h>
#include <stdint.h>

// Problem sizes (fixed by the reference)
static constexpr int Bn = 64;
static constexpr int Sn = 512;
static constexpr int Hn = 1024;

typedef __attribute__((ext_vector_type(16))) __bf16 v16bf;
typedef __attribute__((ext_vector_type(8)))  float  v8f;

// ---- workspace layout (bytes) ----
// encB  bf16 [B,S,H]        :       0 .. 67108864
// w2b   bf16 [H,H]          : 67108864 .. 69206016
// hb    f32  [B,H]          : 69206016 .. 69468160
// scores f32 [B,S]          : 69468160 .. 69599232   (~66.4 MiB total)
static constexpr size_t OFF_W2B    = 67108864;
static constexpr size_t OFF_HB     = 69206016;
static constexpr size_t OFF_SCORES = 69468160;

__device__ __forceinline__ __bf16 f2bf(float f) {
    union { float f; uint32_t u; } s; s.f = f;
    uint32_t r = s.u + 0x7FFFu + ((s.u >> 16) & 1u);   // round-to-nearest-even
    union { unsigned short s; __bf16 b; } o; o.s = (unsigned short)(r >> 16);
    return o.b;
}

// branch-free fast tanh: tanh(x) = 1 - 2/(exp(2x)+1)
// saturates correctly at +/-1 for large |x|; ~4 VALU ops, no EXEC divergence.
__device__ __forceinline__ float fast_tanh(float x) {
    float t = __expf(2.0f * x);
    return 1.0f - 2.0f * __builtin_amdgcn_rcpf(t + 1.0f);
}

// ---- prep: f32 -> bf16 copy of encoder_outputs ----
__global__ __launch_bounds__(256) void cvt_enc(const float* __restrict__ in,
                                               __bf16* __restrict__ out) {
    int i = blockIdx.x * 256 + threadIdx.x;            // one float4 per thread
    if (i >= (Bn * Sn * Hn) / 4) return;
    float4 f = ((const float4*)in)[i];
    union { __bf16 b[4]; uint2 u; } t;
    t.b[0] = f2bf(f.x); t.b[1] = f2bf(f.y); t.b[2] = f2bf(f.z); t.b[3] = f2bf(f.w);
    ((uint2*)out)[i] = t.u;
}

// ---- prep: extract W2 = W[:, H:2H] (row-major [H_out, K]) as bf16 ----
__global__ __launch_bounds__(256) void cvt_w2(const float* __restrict__ W,
                                              __bf16* __restrict__ w2b) {
    int i = blockIdx.x * 256 + threadIdx.x;            // one float4 per thread
    if (i >= (Hn * Hn) / 4) return;
    int e = i * 4;
    int h = e >> 10;
    int k = e & (Hn - 1);
    float4 f = *(const float4*)(W + (size_t)h * (2 * Hn) + Hn + k);
    union { __bf16 b[4]; uint2 u; } t;
    t.b[0] = f2bf(f.x); t.b[1] = f2bf(f.y); t.b[2] = f2bf(f.z); t.b[3] = f2bf(f.w);
    ((uint2*)w2b)[i] = t.u;
}

// ---- prep: hb[b,h] = bias[h] + sum_k hidden[b,k] * W[h, k]  (k < H) ----
__global__ __launch_bounds__(256) void compute_hb(const float* __restrict__ hidden,
                                                  const float* __restrict__ W,
                                                  const float* __restrict__ bias,
                                                  float* __restrict__ hb) {
    __shared__ float sh[Hn];
    int b = blockIdx.x >> 2;
    int h = ((blockIdx.x & 3) << 8) + threadIdx.x;
    for (int k = threadIdx.x; k < Hn; k += 256) sh[k] = hidden[b * Hn + k];
    __syncthreads();
    const float4* wr = (const float4*)(W + (size_t)h * (2 * Hn));
    float acc = bias[h];
#pragma unroll 4
    for (int k4 = 0; k4 < Hn / 4; ++k4) {
        float4 w = wr[k4];
        acc += sh[4 * k4 + 0] * w.x + sh[4 * k4 + 1] * w.y +
               sh[4 * k4 + 2] * w.z + sh[4 * k4 + 3] * w.w;
    }
    hb[b * Hn + h] = acc;
}

// ---- main: fused  scores[b,s] = sum_h tanh( (enc@W2^T)[b,s,h] + hb[b,h] ) * v[h]
// block = 8 waves = one (b, 32-row s-tile). Each wave owns 8 n-tiles (2 groups
// of 4) and both 16-row s-subtiles, so each B fragment feeds 2 WMMAs.
__global__ __launch_bounds__(256) void attn_scores_wmma(
    const __bf16* __restrict__ encB, const __bf16* __restrict__ w2b,
    const float* __restrict__ hb, const float* __restrict__ vvec,
    float* __restrict__ scores) {
    __shared__ float ssum[32];
    const int bi   = blockIdx.x >> 4;            // batch
    const int s0   = (blockIdx.x & 15) << 5;     // s-tile origin (32 rows)
    const int lane = threadIdx.x & 31;
    const int wave = threadIdx.x >> 5;
    const int ln   = lane & 15;                  // row of A / col of B,C
    const int kh   = lane >> 4;                  // K-half selector

    if (threadIdx.x < 32) ssum[threadIdx.x] = 0.0f;
    __syncthreads();

    // A fragment sources: enc rows (s0+ln) and (s0+16+ln), K offset kh*8
    const __bf16* arow0 = encB + ((size_t)(bi * Sn + s0 + ln)) * Hn + kh * 8;
    const __bf16* arow1 = arow0 + (size_t)16 * Hn;

    float part[2][8];
#pragma unroll
    for (int t = 0; t < 2; ++t)
#pragma unroll
        for (int r = 0; r < 8; ++r) part[t][r] = 0.0f;

#pragma unroll
    for (int g = 0; g < 2; ++g) {
        const int nt0 = (wave * 8 + g * 4) * 16;   // first of 4 n-tiles
        v8f c[2][4];
#pragma unroll
        for (int t = 0; t < 2; ++t)
#pragma unroll
            for (int j = 0; j < 4; ++j) c[t][j] = (v8f){0, 0, 0, 0, 0, 0, 0, 0};
        const __bf16* brow[4];
#pragma unroll
        for (int j = 0; j < 4; ++j)
            brow[j] = w2b + ((size_t)(nt0 + j * 16 + ln)) * Hn + kh * 8;

#pragma unroll 2
        for (int k0 = 0; k0 < Hn; k0 += 32) {
            union { v16bf v; uint4 q[2]; } a0, a1;
            a0.q[0] = *(const uint4*)(arow0 + k0);        // K = k0+kh*8 + 0..7
            a0.q[1] = *(const uint4*)(arow0 + k0 + 16);   // K = k0+kh*8 + 16..23
            a1.q[0] = *(const uint4*)(arow1 + k0);
            a1.q[1] = *(const uint4*)(arow1 + k0 + 16);
#pragma unroll
            for (int j = 0; j < 4; ++j) {
                union { v16bf v; uint4 q[2]; } bf_;
                bf_.q[0] = *(const uint4*)(brow[j] + k0);      // K = k0+kh*8+0..7
                bf_.q[1] = *(const uint4*)(brow[j] + k0 + 16); // K = k0+kh*8+16..23
                c[0][j] = __builtin_amdgcn_wmma_f32_16x16x32_bf16(
                    false, a0.v, false, bf_.v, (short)0, c[0][j], false, false);
                c[1][j] = __builtin_amdgcn_wmma_f32_16x16x32_bf16(
                    false, a1.v, false, bf_.v, (short)0, c[1][j], false, false);
            }
        }
        // fused epilogue: tanh(+hb) then dot with v along n
#pragma unroll
        for (int j = 0; j < 4; ++j) {
            const int n = nt0 + j * 16 + ln;
            const float hv = hb[bi * Hn + n];
            const float vv = vvec[n];
#pragma unroll
            for (int t = 0; t < 2; ++t)
#pragma unroll
            for (int r = 0; r < 8; ++r) {
                float e = fast_tanh(c[t][j][r] + hv);
                part[t][r] += e * vv;    // score partial for s-row t*16 + r + 8*kh
            }
        }
    }

    // reduce over the 16 n-lanes of each half, then across waves via LDS
#pragma unroll
    for (int t = 0; t < 2; ++t)
#pragma unroll
    for (int r = 0; r < 8; ++r) {
        float x = part[t][r];
        x += __shfl_xor(x, 1);
        x += __shfl_xor(x, 2);
        x += __shfl_xor(x, 4);
        x += __shfl_xor(x, 8);
        if (ln == 0) atomicAdd(&ssum[t * 16 + r + 8 * kh], x);
    }
    __syncthreads();
    if (threadIdx.x < 32)
        scores[bi * Sn + s0 + threadIdx.x] = ssum[threadIdx.x];
}

// ---- softmax over S per batch row ----
__global__ __launch_bounds__(512) void softmax_rows(const float* __restrict__ scores,
                                                    float* __restrict__ attn) {
    __shared__ float red[512];
    int b = blockIdx.x, t = threadIdx.x;
    float x = scores[b * Sn + t];
    red[t] = x;
    __syncthreads();
    for (int off = 256; off > 0; off >>= 1) {
        if (t < off) red[t] = fmaxf(red[t], red[t + off]);
        __syncthreads();
    }
    float mx = red[0];
    __syncthreads();
    float e = __expf(x - mx);
    red[t] = e;
    __syncthreads();
    for (int off = 256; off > 0; off >>= 1) {
        if (t < off) red[t] += red[t + off];
        __syncthreads();
    }
    attn[b * Sn + t] = e * (1.0f / red[0]);
}

// ---- context[b,h] = sum_s attn[b,s] * enc[b,s,h]  (f32, L2-resident re-read) ----
__global__ __launch_bounds__(256) void context_kernel(const float* __restrict__ enc,
                                                      const float* __restrict__ attn,
                                                      float* __restrict__ ctx) {
    __shared__ float sa[Sn];
    int b = blockIdx.x >> 2;
    int h = ((blockIdx.x & 3) << 8) + threadIdx.x;
    for (int s = threadIdx.x; s < Sn; s += 256) sa[s] = attn[b * Sn + s];
    __syncthreads();
    const float* e = enc + (size_t)b * Sn * Hn + h;
    float acc = 0.0f;
#pragma unroll 4
    for (int s = 0; s < Sn; ++s) acc += sa[s] * e[(size_t)s * Hn];
    ctx[b * Hn + h] = acc;
}

extern "C" void kernel_launch(void* const* d_in, const int* in_sizes, int n_in,
                              void* d_out, int out_size, void* d_ws, size_t ws_size,
                              hipStream_t stream) {
    const float* hidden = (const float*)d_in[0];   // [B,H]
    const float* enc    = (const float*)d_in[1];   // [B,S,H]
    const float* W      = (const float*)d_in[2];   // [H,2H]
    const float* bias   = (const float*)d_in[3];   // [H]
    const float* vvec   = (const float*)d_in[4];   // [H]

    char* ws = (char*)d_ws;                        // needs ~66.4 MiB
    __bf16* encB   = (__bf16*)(ws);
    __bf16* w2b    = (__bf16*)(ws + OFF_W2B);
    float*  hb     = (float*)(ws + OFF_HB);
    float*  scores = (float*)(ws + OFF_SCORES);

    float* ctx  = (float*)d_out;                   // context: 64*1024 floats
    float* attn = (float*)d_out + Bn * Hn;         // attn:    64*512 floats

    cvt_enc<<<(Bn * Sn * Hn) / 4 / 256, 256, 0, stream>>>(enc, encB);
    cvt_w2<<<(Hn * Hn) / 4 / 256, 256, 0, stream>>>(W, w2b);
    compute_hb<<<Bn * (Hn / 256), 256, 0, stream>>>(hidden, W, bias, hb);
    attn_scores_wmma<<<Bn * (Sn / 32), 256, 0, stream>>>(encB, w2b, hb, vvec, scores);
    softmax_rows<<<Bn, 512, 0, stream>>>(scores, attn);
    context_kernel<<<Bn * (Hn / 256), 256, 0, stream>>>(enc, attn, ctx);
}
